// AdapterMLP_91096256348252
// MI455X (gfx1250) — matrix operand order repaired and checked by reading, not compiled, not run on hardware
//
#include <hip/hip_runtime.h>

// ---------------- problem constants ----------------
#define B_   4
#define S_   2048
#define D_   4096
#define W_   128
#define E_   8
#define T_   4
#define KD_  100
#define KDP  128          // KD padded to multiple of 32
#define KI_  1024
#define KMLP 4224         // D_ + KDP
#define NROW (B_ * S_)    // 8192
#define NWORD (B_ * W_)   // 512
#define NENT (NWORD * 9)  // 4608

typedef __attribute__((ext_vector_type(16))) __bf16 v16bf;
typedef __attribute__((ext_vector_type(8)))  float  v8f;

// ---------------- small helpers ----------------
__device__ inline unsigned short f32_to_bf16(float f) {
    unsigned int u = __float_as_uint(f);
    unsigned int r = u + 0x7FFFu + ((u >> 16) & 1u);   // round-to-nearest-even
    return (unsigned short)(r >> 16);
}
__device__ inline float bf16_to_f32(unsigned short h) {
    return __uint_as_float(((unsigned int)h) << 16);
}
__device__ inline v8f zero8() {
    v8f z = {0.f, 0.f, 0.f, 0.f, 0.f, 0.f, 0.f, 0.f};
    return z;
}
__device__ inline float silu(float x) { return x / (1.f + __expf(-x)); }

// ---- WMMA fragment loaders (CDNA5 wave32 layouts, CDNA5 ISA §7.12.2) ----
// A: 16x32 bf16, row-major [M,K]: lane holds row M=lane&15; K chunks at k0+hi*8 and +16.
__device__ inline v16bf load_a_frag(const unsigned short* A, int lda, int m0, int k0, int lane) {
    int m  = m0 + (lane & 15);
    int hi = lane >> 4;
    const unsigned short* p = A + (size_t)m * lda + k0 + hi * 8;
    union { v16bf v; uint4 q[2]; } f;
    f.q[0] = *(const uint4*)(p);
    f.q[1] = *(const uint4*)(p + 16);
    return f.v;
}
// B: 32x16 (KxN); weights row-major [N,K] so lane takes N=lane&15; K chunks at k0+hi*8 and +16.
__device__ inline v16bf load_b_frag(const unsigned short* Bw, int ldb, int n0, int k0, int lane) {
    int n  = n0 + (lane & 15);
    int hi = lane >> 4;
    const unsigned short* p = Bw + (size_t)n * ldb + k0 + hi * 8;
    union { v16bf v; uint4 q[2]; } f;
    f.q[0] = *(const uint4*)(p);
    f.q[1] = *(const uint4*)(p + 16);
    return f.v;
}
// A fragment out of an LDS-staged 32x64 tile (row stride 64 elements).
__device__ inline v16bf load_a_lds(const unsigned short* sbuf, int i16, int ks, int lane) {
    int m  = i16 * 16 + (lane & 15);
    int hi = lane >> 4;
    const unsigned short* p = sbuf + m * 64 + ks + hi * 8;
    union { v16bf v; uint4 q[2]; } f;
    f.q[0] = *(const uint4*)(p);
    f.q[1] = *(const uint4*)(p + 16);
    return f.v;
}

// ---------------- conversion kernels ----------------
__global__ void cvt_pad_bf16(const float* __restrict__ src, unsigned short* __restrict__ dst,
                             int rows, int scols, int dcols) {
    size_t total = (size_t)rows * dcols;
    for (size_t i = blockIdx.x * (size_t)blockDim.x + threadIdx.x; i < total;
         i += (size_t)gridDim.x * blockDim.x) {
        int r = (int)(i / dcols), c = (int)(i % dcols);
        float v = (c < scols) ? src[(size_t)r * scols + c] : 0.f;
        dst[i] = f32_to_bf16(v);
    }
}

__global__ void cvt_transpose_bf16(const float* __restrict__ src, unsigned short* __restrict__ dst,
                                   int R, int C) {  // src [R,C] -> dst [C,R]
    size_t total = (size_t)R * C;
    for (size_t i = blockIdx.x * (size_t)blockDim.x + threadIdx.x; i < total;
         i += (size_t)gridDim.x * blockDim.x) {
        int r = (int)(i / C), c = (int)(i % C);
        dst[(size_t)c * R + r] = f32_to_bf16(src[i]);
    }
}

// ---------------- RMSNorm -> bf16 ----------------
__global__ void rmsnorm_kernel(const float* __restrict__ x, const float* __restrict__ w,
                               unsigned short* __restrict__ out) {
    int row = blockIdx.x;                       // NROW rows
    const float* xr = x + (size_t)row * D_;
    __shared__ float red[256];
    float s = 0.f;
    for (int i = threadIdx.x; i < D_; i += blockDim.x) { float v = xr[i]; s += v * v; }
    red[threadIdx.x] = s;
    __syncthreads();
    for (int off = 128; off > 0; off >>= 1) {
        if ((int)threadIdx.x < off) red[threadIdx.x] += red[threadIdx.x + off];
        __syncthreads();
    }
    float rs = rsqrtf(red[0] / (float)D_ + 1e-6f);
    for (int i = threadIdx.x; i < D_; i += blockDim.x)
        out[(size_t)row * D_ + i] = f32_to_bf16(w[i] * xr[i] * rs);
}

// ---------------- subtoken pooling: b[w] = mean of valid normed rows ----------------
__global__ void word_pool_kernel(const unsigned short* __restrict__ normed,
                                 const int* __restrict__ ws_idx,
                                 unsigned short* __restrict__ b_bf16) {
    int word = blockIdx.x;                      // NWORD
    int bi = word / W_;
    int idxs[T_]; int cnt = 0;
    for (int t = 0; t < T_; ++t) {
        idxs[t] = ws_idx[word * T_ + t];
        if (idxs[t] != -1) ++cnt;
    }
    float inv = cnt ? 1.f / (float)cnt : 0.f;
    for (int d = threadIdx.x; d < D_; d += blockDim.x) {
        float s = 0.f;
        for (int t = 0; t < T_; ++t)
            if (idxs[t] != -1)
                s += bf16_to_f32(normed[((size_t)bi * S_ + idxs[t]) * D_ + d]);
        b_bf16[(size_t)word * D_ + d] = f32_to_bf16(s * inv);
    }
}

// ---------------- gather entity embeddings + sentinel ----------------
__global__ void build_ents_kernel(const int* __restrict__ we, const float* __restrict__ cembed,
                                  const float* __restrict__ sentinel,
                                  unsigned short* __restrict__ ent_bf16,
                                  float* __restrict__ ent_f32) {
    int row = blockIdx.x;                       // NENT
    int e = row % 9, word = row / 9;
    const float* src;
    if (e == 8) src = sentinel;
    else { int id = we[word * E_ + e]; if (id < 0) id = 0; src = cembed + (size_t)id * KD_; }
    for (int k = threadIdx.x; k < KDP; k += blockDim.x) {
        float v = (k < KD_) ? src[k] : 0.f;
        ent_f32[(size_t)row * KDP + k] = v;
        ent_bf16[(size_t)row * KDP + k] = f32_to_bf16(v);
    }
}

// ---------------- fused gate/up WMMA GEMM: h = silu(X@Gw^T) * (X@Uw^T), bf16 out ----------
__global__ void wmma_gateup_kernel(const unsigned short* __restrict__ X,
                                   const unsigned short* __restrict__ Gw,
                                   const unsigned short* __restrict__ Uw,
                                   unsigned short* __restrict__ H) {
    int wave = (int)((blockIdx.x * blockDim.x + threadIdx.x) >> 5);
    int lane = threadIdx.x & 31;
    const int tiles_n = KI_ / 64;               // 16
    int m0 = (wave / tiles_n) * 16;
    int n0 = (wave % tiles_n) * 64;
    v8f ag[4], au[4];
    for (int j = 0; j < 4; ++j) { ag[j] = zero8(); au[j] = zero8(); }
    for (int k0 = 0; k0 < KDP; k0 += 32) {
        v16bf a = load_a_frag(X, KDP, m0, k0, lane);
        for (int j = 0; j < 4; ++j) {
            v16bf bg = load_b_frag(Gw, KDP, n0 + j * 16, k0, lane);
            ag[j] = __builtin_amdgcn_wmma_f32_16x16x32_bf16(false, a, false, bg, (short)0, ag[j], false, false);
            v16bf bu = load_b_frag(Uw, KDP, n0 + j * 16, k0, lane);
            au[j] = __builtin_amdgcn_wmma_f32_16x16x32_bf16(false, a, false, bu, (short)0, au[j], false, false);
        }
    }
    int mb = m0 + (lane >> 4) * 8;
    for (int j = 0; j < 4; ++j) {
        int n = n0 + j * 16 + (lane & 15);
        for (int r = 0; r < 8; ++r) {
            float g = ag[j][r], u = au[j][r];
            H[(size_t)(mb + r) * KI_ + n] = f32_to_bf16(silu(g) * u);
        }
    }
}

// ---------------- generic WMMA GEMM: C[M,N] = A[M,K] * Bw[N,K]^T, fp32 out -------------
__global__ void wmma_gemm_nt_kernel(const unsigned short* __restrict__ A,
                                    const unsigned short* __restrict__ Bw,
                                    float* __restrict__ C, int M, int N, int K) {
    int wave = (int)((blockIdx.x * blockDim.x + threadIdx.x) >> 5);
    int lane = threadIdx.x & 31;
    int tiles_n = N / 64;
    int m0 = (wave / tiles_n) * 32;
    int n0 = (wave % tiles_n) * 64;
    v8f acc[2][4];
    for (int i = 0; i < 2; ++i)
        for (int j = 0; j < 4; ++j) acc[i][j] = zero8();
    for (int k0 = 0; k0 < K; k0 += 32) {
        v16bf a0 = load_a_frag(A, K, m0,      k0, lane);
        v16bf a1 = load_a_frag(A, K, m0 + 16, k0, lane);
        for (int j = 0; j < 4; ++j) {
            v16bf b = load_b_frag(Bw, K, n0 + j * 16, k0, lane);
            acc[0][j] = __builtin_amdgcn_wmma_f32_16x16x32_bf16(false, a0, false, b, (short)0, acc[0][j], false, false);
            acc[1][j] = __builtin_amdgcn_wmma_f32_16x16x32_bf16(false, a1, false, b, (short)0, acc[1][j], false, false);
        }
    }
    for (int i = 0; i < 2; ++i) {
        int mb = m0 + i * 16 + (lane >> 4) * 8;
        for (int j = 0; j < 4; ++j) {
            int n = n0 + j * 16 + (lane & 15);
            for (int r = 0; r < 8; ++r)
                C[(size_t)(mb + r) * N + n] = acc[i][j][r];
        }
    }
}

// ---------------- attention ----------------
__global__ void attn_kernel(const unsigned short* __restrict__ h, const float* __restrict__ bd,
                            const int* __restrict__ we, const float* __restrict__ ent_f32,
                            float* __restrict__ attn_out) {
    int word = blockIdx.x;                      // NWORD; blockDim = 288 (9 waves)
    int e = threadIdx.x >> 5;
    int lane = threadIdx.x & 31;
    __shared__ float aw[9];
    __shared__ float attn[9];
    const unsigned short* hr = h + ((size_t)word * 9 + e) * KI_;
    const float* bdr = bd + (size_t)word * KI_;
    float s = 0.f;
    for (int i = lane; i < KI_; i += 32) s += bf16_to_f32(hr[i]) * bdr[i];
    for (int off = 16; off > 0; off >>= 1) s += __shfl_xor(s, off, 32);
    if (lane == 0) aw[e] = s;
    __syncthreads();
    if (threadIdx.x == 0) {
        float vals[9]; float m = -1e30f;
        for (int q = 0; q < 9; ++q) {
            float msk = (q < 8 && we[word * E_ + q] == -1) ? -1e9f : 0.f;
            vals[q] = aw[q] + msk;
            m = fmaxf(m, vals[q]);
        }
        float sum = 0.f;
        for (int q = 0; q < 9; ++q) { vals[q] = __expf(vals[q] - m); sum += vals[q]; }
        float inv = 1.f / sum;
        for (int q = 0; q < 9; ++q) attn[q] = vals[q] * inv;
    }
    __syncthreads();
    for (int k = threadIdx.x; k < KDP; k += blockDim.x) {
        float s2 = 0.f;
        for (int q = 0; q < 9; ++q)
            s2 += attn[q] * ent_f32[((size_t)word * 9 + q) * KDP + k];
        attn_out[(size_t)word * KDP + k] = s2;
    }
}

// ---------------- zero + scatter infusion rows ----------------
__global__ void zero_u32_kernel(unsigned int* __restrict__ p, size_t n) {
    for (size_t i = blockIdx.x * (size_t)blockDim.x + threadIdx.x; i < n;
         i += (size_t)gridDim.x * blockDim.x) p[i] = 0u;
}

__global__ void scatter_kernel(const float* __restrict__ attn_out, const int* __restrict__ ws_idx,
                               unsigned short* __restrict__ tmp) {
    int pair = blockIdx.x;                      // NWORD * T_
    int word = pair / T_;
    int bi = word / W_;
    int sidx = ws_idx[pair];
    if (sidx < 0) return;
    unsigned short* dst = tmp + ((size_t)bi * S_ + sidx) * KDP;
    for (int k = threadIdx.x; k < KDP; k += blockDim.x)
        dst[k] = f32_to_bf16(attn_out[(size_t)word * KDP + k]);
}

// ================= big fused WMMA GEMM with async-LDS A staging =================
// out = silu([Xn|Xt] @ Wm^T + bias) * alpha + residual.
// Block = 8 waves sharing one 32-row M tile; each block stages its 32x64 A chunk
// into LDS with GLOBAL_LOAD_ASYNC_TO_LDS_B128 (ASYNCcnt, double buffered),
// while each wave streams its own 64-column B panel from L2.
#define NSTAGE ((D_ + KDP) / 64)    // 66 chunks of K=64

__device__ inline void stage_A(const unsigned short* __restrict__ Xn,
                               const unsigned short* __restrict__ Xt,
                               unsigned short* sA, int buf, int s, int m0, int tid) {
    const unsigned short* src; int lda, kb;
    if (s < D_ / 64) { src = Xn; lda = D_;  kb = s * 64; }
    else             { src = Xt; lda = KDP; kb = (s - D_ / 64) * 64; }
    int row = tid >> 3;                  // 32 rows, 8 chunks of 16B per 128B row
    int col = (tid & 7) * 8;             // element offset (8 bf16 = 16B)
    const unsigned short* g = src + (size_t)(m0 + row) * lda + kb + col;
    // low 32 bits of the flat address of a shared object == wave-relative LDS offset
    unsigned ldsa = (unsigned)(size_t)(const void*)(sA + buf * (32 * 64) + row * 64 + col);
    asm volatile("global_load_async_to_lds_b128 %0, %1, off"
                 :: "v"(ldsa), "v"(g) : "memory");
}

__global__ void wmma_final_kernel(const unsigned short* __restrict__ Xn,   // [NROW, D_]
                                  const unsigned short* __restrict__ Xt,   // [NROW, KDP]
                                  const unsigned short* __restrict__ Wm,   // [D_, KMLP]
                                  const float* __restrict__ bias,
                                  const float* __restrict__ residual,
                                  const float* __restrict__ alpha,
                                  float* __restrict__ out) {
    __shared__ unsigned short sA[2][32 * 64];   // double-buffered 32x64 bf16 A chunk (8 KB)
    const int tiles_n = D_ / 64;                // 64 (multiple of 8 => m0 uniform per block)
    int tid  = threadIdx.x;
    int lane = tid & 31;
    int wib  = tid >> 5;                        // wave in block, 0..7
    int wave0 = blockIdx.x * 8;
    int m0 = (wave0 / tiles_n) * 32;            // uniform across block
    int n0 = ((wave0 + wib) % tiles_n) * 64;

    v8f acc[2][4];
    for (int i = 0; i < 2; ++i)
        for (int j = 0; j < 4; ++j) acc[i][j] = zero8();

    stage_A(Xn, Xt, &sA[0][0], 0, 0, m0, tid);          // prologue: chunk 0

    for (int s = 0; s < NSTAGE; ++s) {
        int buf = s & 1;
        if (s + 1 < NSTAGE) {
            stage_A(Xn, Xt, &sA[0][0], buf ^ 1, s + 1, m0, tid);
            asm volatile("s_wait_asynccnt 0x1" ::: "memory");   // chunk s landed
        } else {
            asm volatile("s_wait_asynccnt 0x0" ::: "memory");
        }
        __syncthreads();

        const unsigned short* sbuf = &sA[buf][0];
        int kw = s * 64;                                   // K offset into Wm
        // issue all loads of this chunk up front (distinct regs), then 16 WMMAs
        v16bf a0A = load_a_lds(sbuf, 0, 0,  lane);
        v16bf a1A = load_a_lds(sbuf, 1, 0,  lane);
        v16bf a0B = load_a_lds(sbuf, 0, 32, lane);
        v16bf a1B = load_a_lds(sbuf, 1, 32, lane);
        v16bf bA0 = load_b_frag(Wm, KMLP, n0,      kw,      lane);
        v16bf bA1 = load_b_frag(Wm, KMLP, n0 + 16, kw,      lane);
        v16bf bA2 = load_b_frag(Wm, KMLP, n0 + 32, kw,      lane);
        v16bf bA3 = load_b_frag(Wm, KMLP, n0 + 48, kw,      lane);
        v16bf bB0 = load_b_frag(Wm, KMLP, n0,      kw + 32, lane);
        v16bf bB1 = load_b_frag(Wm, KMLP, n0 + 16, kw + 32, lane);
        v16bf bB2 = load_b_frag(Wm, KMLP, n0 + 32, kw + 32, lane);
        v16bf bB3 = load_b_frag(Wm, KMLP, n0 + 48, kw + 32, lane);

        acc[0][0] = __builtin_amdgcn_wmma_f32_16x16x32_bf16(false, a0A, false, bA0, (short)0, acc[0][0], false, false);
        acc[1][0] = __builtin_amdgcn_wmma_f32_16x16x32_bf16(false, a1A, false, bA0, (short)0, acc[1][0], false, false);
        acc[0][1] = __builtin_amdgcn_wmma_f32_16x16x32_bf16(false, a0A, false, bA1, (short)0, acc[0][1], false, false);
        acc[1][1] = __builtin_amdgcn_wmma_f32_16x16x32_bf16(false, a1A, false, bA1, (short)0, acc[1][1], false, false);
        acc[0][2] = __builtin_amdgcn_wmma_f32_16x16x32_bf16(false, a0A, false, bA2, (short)0, acc[0][2], false, false);
        acc[1][2] = __builtin_amdgcn_wmma_f32_16x16x32_bf16(false, a1A, false, bA2, (short)0, acc[1][2], false, false);
        acc[0][3] = __builtin_amdgcn_wmma_f32_16x16x32_bf16(false, a0A, false, bA3, (short)0, acc[0][3], false, false);
        acc[1][3] = __builtin_amdgcn_wmma_f32_16x16x32_bf16(false, a1A, false, bA3, (short)0, acc[1][3], false, false);
        acc[0][0] = __builtin_amdgcn_wmma_f32_16x16x32_bf16(false, a0B, false, bB0, (short)0, acc[0][0], false, false);
        acc[1][0] = __builtin_amdgcn_wmma_f32_16x16x32_bf16(false, a1B, false, bB0, (short)0, acc[1][0], false, false);
        acc[0][1] = __builtin_amdgcn_wmma_f32_16x16x32_bf16(false, a0B, false, bB1, (short)0, acc[0][1], false, false);
        acc[1][1] = __builtin_amdgcn_wmma_f32_16x16x32_bf16(false, a1B, false, bB1, (short)0, acc[1][1], false, false);
        acc[0][2] = __builtin_amdgcn_wmma_f32_16x16x32_bf16(false, a0B, false, bB2, (short)0, acc[0][2], false, false);
        acc[1][2] = __builtin_amdgcn_wmma_f32_16x16x32_bf16(false, a1B, false, bB2, (short)0, acc[1][2], false, false);
        acc[0][3] = __builtin_amdgcn_wmma_f32_16x16x32_bf16(false, a0B, false, bB3, (short)0, acc[0][3], false, false);
        acc[1][3] = __builtin_amdgcn_wmma_f32_16x16x32_bf16(false, a1B, false, bB3, (short)0, acc[1][3], false, false);

        __syncthreads();   // all waves done reading buf before it is overwritten
    }

    float al = alpha[0];
    for (int i = 0; i < 2; ++i) {
        int mb = m0 + i * 16 + (lane >> 4) * 8;
        for (int j = 0; j < 4; ++j) {
            int n = n0 + j * 16 + (lane & 15);
            float bs = bias[n];
            for (int r = 0; r < 8; ++r) {
                size_t idx = (size_t)(mb + r) * D_ + n;
                float v = acc[i][j][r] + bs;
                out[idx] = silu(v) * al + residual[idx];
            }
        }
    }
}

// ---------------- host launcher ----------------
extern "C" void kernel_launch(void* const* d_in, const int* in_sizes, int n_in,
                              void* d_out, int out_size, void* d_ws, size_t ws_size,
                              hipStream_t stream) {
    const float* hid      = (const float*)d_in[0];   // [B,S,D]
    const int*   we       = (const int*)d_in[1];     // [B,W,E]
    const int*   ws_idx   = (const int*)d_in[2];     // [B,W,T]
    // d_in[3] input_ids unused
    const float* cembed   = (const float*)d_in[4];   // [V,KD]
    const float* sentinel = (const float*)d_in[5];   // [1,KD]
    const float* ln_w     = (const float*)d_in[6];   // [D]
    const float* gate_w   = (const float*)d_in[7];   // [KI,KD]
    const float* up_w     = (const float*)d_in[8];   // [KI,KD]
    const float* down_w   = (const float*)d_in[9];   // [D,KI]
    const float* mlp_w    = (const float*)d_in[10];  // [D,D+KD]
    const float* mlp_b    = (const float*)d_in[11];  // [D]
    const float* alpha    = (const float*)d_in[12];  // [1]
    float* out = (float*)d_out;

    char* ws = (char*)d_ws;
    size_t off = 0;
    auto carve = [&](size_t bytes) -> char* {
        char* p = ws + off;
        off = (off + bytes + 255) & ~(size_t)255;
        return p;
    };
    unsigned short* normed_bf = (unsigned short*)carve((size_t)NROW * D_ * 2);
    unsigned short* mlpw_bf   = (unsigned short*)carve((size_t)D_ * KMLP * 2);
    unsigned short* gw_bf     = (unsigned short*)carve((size_t)KI_ * KDP * 2);
    unsigned short* uw_bf     = (unsigned short*)carve((size_t)KI_ * KDP * 2);
    unsigned short* dwT_bf    = (unsigned short*)carve((size_t)KI_ * D_ * 2);
    unsigned short* ent_bf    = (unsigned short*)carve((size_t)NENT * KDP * 2);
    float*          ent_f     = (float*)carve((size_t)NENT * KDP * 4);
    unsigned short* b_bf      = (unsigned short*)carve((size_t)NWORD * D_ * 2);
    unsigned short* h_bf      = (unsigned short*)carve((size_t)NENT * KI_ * 2);
    float*          bd_f      = (float*)carve((size_t)NWORD * KI_ * 4);
    float*          attn_out  = (float*)carve((size_t)NWORD * KDP * 4);
    unsigned short* tmp_bf    = (unsigned short*)carve((size_t)NROW * KDP * 2);
    (void)ws_size; (void)n_in; (void)in_sizes; (void)out_size;

    // 1) weight conversions to bf16 (padded / transposed)
    cvt_pad_bf16<<<256,  256, 0, stream>>>(gate_w, gw_bf,  KI_, KD_, KDP);
    cvt_pad_bf16<<<256,  256, 0, stream>>>(up_w,   uw_bf,  KI_, KD_, KDP);
    cvt_pad_bf16<<<4096, 256, 0, stream>>>(mlp_w,  mlpw_bf, D_, D_ + KD_, KMLP);
    cvt_transpose_bf16<<<2048, 256, 0, stream>>>(down_w, dwT_bf, D_, KI_);   // -> [KI, D]

    // 2) RMSNorm -> bf16
    rmsnorm_kernel<<<NROW, 256, 0, stream>>>(hid, ln_w, normed_bf);

    // 3) subtoken pooling b[w]
    word_pool_kernel<<<NWORD, 256, 0, stream>>>(normed_bf, ws_idx, b_bf);

    // 4) gather entity embeddings (+ sentinel), padded to 128
    build_ents_kernel<<<NENT, 128, 0, stream>>>(we, cembed, sentinel, ent_bf, ent_f);

    // 5) fused SwiGLU gate/up via WMMA: h = silu(X@G^T)*(X@U^T)   [NENT, KI]
    wmma_gateup_kernel<<<(NENT / 16) * (KI_ / 64) / 8, 256, 0, stream>>>(ent_bf, gw_bf, uw_bf, h_bf);

    // 6) bd = b @ down_w via WMMA  ([NWORD, D] x [KI, D]^T -> [NWORD, KI])
    wmma_gemm_nt_kernel<<<(NWORD / 32) * (KI_ / 64) / 8, 256, 0, stream>>>(b_bf, dwT_bf, bd_f,
                                                                           NWORD, KI_, D_);

    // 7) attention over 9 entities per word -> attn_out [NWORD, KDP]
    attn_kernel<<<NWORD, 288, 0, stream>>>(h_bf, bd_f, we, ent_f, attn_out);

    // 8) zero + scatter infusion rows tmp [NROW, KDP]
    zero_u32_kernel<<<512, 256, 0, stream>>>((unsigned int*)tmp_bf, (size_t)NROW * KDP / 2);
    scatter_kernel<<<NWORD * T_, 128, 0, stream>>>(attn_out, ws_idx, tmp_bf);

    // 9) big fused GEMM + silu + alpha + residual  (M=8192, N=4096, K=4224)
    wmma_final_kernel<<<(NROW / 32) * (D_ / 64) / 8, 256, 0, stream>>>(
        normed_bf, tmp_bf, mlpw_bf, mlp_b, hid, alpha, out);
}
